// HierarchicalAttention_72353019068548
// MI455X (gfx1250) — compile-verified, matrix operand order checked by reading, not run on hardware
//
#include <hip/hip_runtime.h>
#include <hip/hip_bf16.h>

typedef __attribute__((ext_vector_type(16))) __bf16 bf16x16;
typedef __attribute__((ext_vector_type(8)))  __bf16 bf16x8;
typedef __attribute__((ext_vector_type(8)))  float  f32x8;

union BFrag { bf16x16 v; struct { bf16x8 lo, hi; } p; };

// Load a 16-element bf16 A/B fragment for wmma_f32_16x16x32_bf16.
// Per ISA layout: elements 0..7 hold K = hi*8 .. hi*8+7, elements 8..15 hold
// K = 16+hi*8 .. 16+hi*8+7 (caller passes base already offset by hi*8).
__device__ __forceinline__ bf16x16 load_frag(const __bf16* __restrict__ base) {
    BFrag f;
    f.p.lo = *(const bf16x8*)(base);
    f.p.hi = *(const bf16x8*)(base + 16);
    return f.v;
}

__device__ __forceinline__ f32x8 wmma_bf16(bf16x16 a, bf16x16 b, f32x8 c) {
    return __builtin_amdgcn_wmma_f32_16x16x32_bf16(
        /*neg_a=*/false, a, /*neg_b=*/false, b,
        /*c_mod=*/(short)0, c, /*reuse_a=*/false, /*reuse_b=*/false);
}

// CDNA5 async copy: 16 bytes global -> LDS, tracked by ASYNCcnt.
// Flat pointers to LDS keep the workgroup-relative offset in addr[31:0],
// which is exactly what the LDS-address VGPR of the instruction wants.
__device__ __forceinline__ void async_copy_b128(const __bf16* g, __bf16* l) {
    unsigned lds_addr = (unsigned)(unsigned long long)(void*)l;
    asm volatile("global_load_async_to_lds_b128 %0, %1, off"
                 :: "v"(lds_addr), "v"(g) : "memory");
}
__device__ __forceinline__ void wait_async0() {
    asm volatile("s_wait_asynccnt 0x0" ::: "memory");
}

// ---------------------------------------------------------------- cast kernel
__global__ __launch_bounds__(256)
void cast_f32_to_bf16(const float* __restrict__ x, __bf16* __restrict__ y, int n) {
    int i = (blockIdx.x * 256 + threadIdx.x) * 4;
    if (i + 3 < n) {
        float4 v = *(const float4*)(x + i);
        y[i + 0] = (__bf16)v.x;
        y[i + 1] = (__bf16)v.y;
        y[i + 2] = (__bf16)v.z;
        y[i + 3] = (__bf16)v.w;
    } else {
        for (int j = i; j < n; ++j) y[j] = (__bf16)x[j];
    }
}

// --------------------------------------------------------- shared GEMM mainloop
// Block = 128 threads (4 waves). Block tile: 128 rows (32 per wave) x 64 cols
// (shared). B slab (64 rows x 32 k, bf16 = 4KB) is double-buffered in LDS and
// filled with global_load_async_to_lds_b128. Each wave holds 2 A fragments
// (pipelined one k-step ahead) and 8 accumulators, so every B fragment is
// reused twice: 8 WMMAs per slab per wave. K = 1024.
__device__ __forceinline__ void gemm_mainloop(
    const __bf16* __restrict__ X, const __bf16* __restrict__ W,
    int mbase_wave, int nbase, __bf16* lbuf /* [2][64*32] */, f32x8 acc[2][4]) {
    const int K = 1024;
    const int tid  = threadIdx.x;
    const int lane = tid & 31;
    const int hi   = lane >> 4;
    const int ln   = lane & 15;

    const __bf16* arow0 = X + (size_t)(mbase_wave + ln) * K + hi * 8;
    const __bf16* arow1 = arow0 + (size_t)16 * K;
    const __bf16* wbase = W + (size_t)nbase * K;

    // stage one 64x32 B slab into LDS buffer `sel` via async copies:
    // 256 chunks of 16B, 2 per thread.
    auto stage = [&](int sel, int kt) {
        __bf16* dst = lbuf + sel * (64 * 32);
#pragma unroll
        for (int c = tid; c < 256; c += 128) {
            const int row = c >> 2;
            const int ko  = (c & 3) * 8;
            async_copy_b128(wbase + (size_t)row * K + kt + ko, dst + row * 32 + ko);
        }
    };

    bf16x16 a0 = load_frag(arow0);
    bf16x16 a1 = load_frag(arow1);
    stage(0, 0);

    const int steps = K / 32;
    for (int i = 0; i < steps; ++i) {
        wait_async0();
        __syncthreads();                       // slab i&1 visible to all waves
        bf16x16 a0n = a0, a1n = a1;
        if (i + 1 < steps) {
            stage((i + 1) & 1, (i + 1) * 32);  // overlap next slab with compute
            __builtin_prefetch(arow0 + (i + 2) * 32, 0, 3);
            __builtin_prefetch(arow1 + (i + 2) * 32, 0, 3);
            a0n = load_frag(arow0 + (i + 1) * 32);
            a1n = load_frag(arow1 + (i + 1) * 32);
        }
        const __bf16* buf = lbuf + (i & 1) * (64 * 32);
        // Preload all four B fragments into distinct registers so the
        // per-WMMA dscnt waits can stagger instead of draining to zero.
        BFrag b[4];
#pragma unroll
        for (int nt = 0; nt < 4; ++nt) {
            const __bf16* bb = buf + (nt * 16 + ln) * 32 + hi * 8;
            b[nt].p.lo = *(const bf16x8*)bb;        // ds_load_b128
            b[nt].p.hi = *(const bf16x8*)(bb + 16); // ds_load_b128
        }
#pragma unroll
        for (int nt = 0; nt < 4; ++nt) {
            acc[0][nt] = wmma_bf16(a0, b[nt].v, acc[0][nt]);
            acc[1][nt] = wmma_bf16(a1, b[nt].v, acc[1][nt]);
        }
        a0 = a0n;
        a1 = a1n;
    }
}

// ------------------------------------------------------------- QKV projection
// C[m,n] = sum_k X[m,k] * W[n,k] + bias[n],  M=8192, N=3072, K=1024.
// Epilogue scatters into per-(window,head) Q [W,64], K [W,64], and V^T [64,W].
__global__ __launch_bounds__(128)
void gemm_qkv(const __bf16* __restrict__ X, const __bf16* __restrict__ W,
              const float* __restrict__ bias,
              __bf16* __restrict__ Qb, __bf16* __restrict__ Kb,
              __bf16* __restrict__ Vt,
              int wshift /* log2(window) */, float qscale) {
    __shared__ __align__(16) __bf16 lbuf[2 * 64 * 32];
    const int lane = threadIdx.x & 31;
    const int wv   = threadIdx.x >> 5;
    const int hi   = lane >> 4;
    const int ln   = lane & 15;
    const int window = 1 << wshift;
    const int nstrips = 3072 / 64;
    const int bm = blockIdx.x / nstrips;
    const int bn = blockIdx.x % nstrips;
    const int mbase = bm * 128 + wv * 32;
    const int nbase = bn * 64;

    f32x8 acc[2][4] = {};
    gemm_mainloop(X, W, mbase, nbase, lbuf, acc);

#pragma unroll
    for (int nt = 0; nt < 4; ++nt) {
        const int n   = nbase + nt * 16 + ln;
        const float bv = bias[n];
        const int sec = n >> 10;          // 0=Q 1=K 2=V
        const int h   = (n & 1023) >> 6;  // head
        const int d   = n & 63;           // dim within head
#pragma unroll
        for (int mt = 0; mt < 2; ++mt) {
#pragma unroll
            for (int r = 0; r < 8; ++r) {
                const int m    = mbase + mt * 16 + r + 8 * hi;
                const int win  = m >> wshift;
                const int qpos = m & (window - 1);
                const float v  = acc[mt][nt][r] + bv;
                const int base = win * 16 + h;
                if (sec == 0)
                    Qb[((size_t)base * window + qpos) * 64 + d] = (__bf16)(v * qscale);
                else if (sec == 1)
                    Kb[((size_t)base * window + qpos) * 64 + d] = (__bf16)v;
                else
                    Vt[((size_t)base * 64 + d) * window + qpos] = (__bf16)v;
            }
        }
    }
}

// ----------------------------------------------------------------- attention
// One wave = one 16-query tile of one head of one window. Flash-style online
// softmax over KV chunks of 32 columns. All K and V^T fragments of a chunk are
// issued before the first WMMA so a single loadcnt wait covers the batch.
__global__ __launch_bounds__(128)
void attn_kernel(const __bf16* __restrict__ Qb, const __bf16* __restrict__ Kb,
                 const __bf16* __restrict__ Vt, __bf16* __restrict__ O,
                 int wshift) {
    __shared__ __align__(16) __bf16 pbuf[4][16 * 32];

    const int window = 1 << wshift;
    const int lane = threadIdx.x & 31;
    const int wv   = threadIdx.x >> 5;
    const int hi   = lane >> 4;
    const int ln   = lane & 15;
    const int qtshift = wshift - 4;

    const int gw  = blockIdx.x * 4 + wv;
    const int qt  = gw & ((1 << qtshift) - 1);
    const int t1  = gw >> qtshift;
    const int h   = t1 & 15;
    const int win = t1 >> 4;

    const __bf16* qbase = Qb + ((size_t)(win * 16 + h) * window) * 64;
    const __bf16* kbase = Kb + ((size_t)(win * 16 + h) * window) * 64;
    const __bf16* vbase = Vt + ((size_t)(win * 16 + h) * 64) * window;

    const __bf16* qrow = qbase + (size_t)(qt * 16 + ln) * 64 + hi * 8;
    const bf16x16 qa0 = load_frag(qrow);
    const bf16x16 qa1 = load_frag(qrow + 32);

    f32x8 o[4] = {};
    float mrow[8], lrow[8];
#pragma unroll
    for (int r = 0; r < 8; ++r) { mrow[r] = -3.0e38f; lrow[r] = 0.0f; }

    __bf16* pb = pbuf[wv];

    for (int c = 0; c < window; c += 32) {
        // ---- batch all loads for this chunk first
        const __bf16* kr0 = kbase + (size_t)(c + ln) * 64 + hi * 8;
        const bf16x16 k00 = load_frag(kr0);
        const bf16x16 k01 = load_frag(kr0 + 32);
        const bf16x16 k10 = load_frag(kr0 + 16 * 64);
        const bf16x16 k11 = load_frag(kr0 + 16 * 64 + 32);
        bf16x16 vb[4];
#pragma unroll
        for (int t = 0; t < 4; ++t)
            vb[t] = load_frag(vbase + (size_t)(t * 16 + ln) * window + c + hi * 8);

        // ---- scores (scale pre-baked into Q)
        f32x8 s0 = {}, s1 = {};
        s0 = wmma_bf16(qa0, k00, s0);
        s0 = wmma_bf16(qa1, k01, s0);
        s1 = wmma_bf16(qa0, k10, s1);
        s1 = wmma_bf16(qa1, k11, s1);

        // ---- online softmax (rows r+8*hi live across the 16 lanes of a half)
        float f[8];
#pragma unroll
        for (int r = 0; r < 8; ++r) {
            float t = fmaxf(s0[r], s1[r]);
            t = fmaxf(t, __shfl_xor(t, 1, 32));
            t = fmaxf(t, __shfl_xor(t, 2, 32));
            t = fmaxf(t, __shfl_xor(t, 4, 32));
            t = fmaxf(t, __shfl_xor(t, 8, 32));
            const float mn = fmaxf(mrow[r], t);
            const float p0 = __expf(s0[r] - mn);
            const float p1 = __expf(s1[r] - mn);
            const float fs = __expf(mrow[r] - mn);
            mrow[r] = mn;
            float rs = p0 + p1;
            rs += __shfl_xor(rs, 1, 32);
            rs += __shfl_xor(rs, 2, 32);
            rs += __shfl_xor(rs, 4, 32);
            rs += __shfl_xor(rs, 8, 32);
            lrow[r] = lrow[r] * fs + rs;
            f[r] = fs;
            pb[(r + 8 * hi) * 32 + ln]      = (__bf16)p0;
            pb[(r + 8 * hi) * 32 + 16 + ln] = (__bf16)p1;
        }
#pragma unroll
        for (int t = 0; t < 4; ++t)
#pragma unroll
            for (int r = 0; r < 8; ++r) o[t][r] *= f[r];

        // ---- re-stripe P (16x32) into A-fragment layout via LDS
        BFrag pf;
        pf.p.lo = *(const bf16x8*)&pb[ln * 32 + hi * 8];
        pf.p.hi = *(const bf16x8*)&pb[ln * 32 + 16 + hi * 8];

        // ---- PV accumulate
#pragma unroll
        for (int t = 0; t < 4; ++t) o[t] = wmma_bf16(pf.v, vb[t], o[t]);
    }

    // ---- epilogue: normalize, store bf16 row-major [B*S, 1024]
#pragma unroll
    for (int t = 0; t < 4; ++t) {
        const int col = h * 64 + t * 16 + ln;
#pragma unroll
        for (int r = 0; r < 8; ++r) {
            const int row = win * window + qt * 16 + r + 8 * hi;
            O[(size_t)row * 1024 + col] = (__bf16)(o[t][r] / lrow[r]);
        }
    }
}

// ----------------------------------------------------------- output projection
__global__ __launch_bounds__(128)
void gemm_out(const __bf16* __restrict__ A, const __bf16* __restrict__ W,
              const float* __restrict__ bias, float* __restrict__ C) {
    __shared__ __align__(16) __bf16 lbuf[2 * 64 * 32];
    const int N = 1024;
    const int lane = threadIdx.x & 31;
    const int wv   = threadIdx.x >> 5;
    const int hi   = lane >> 4;
    const int ln   = lane & 15;
    const int nstrips = N / 64;
    const int bm = blockIdx.x / nstrips;
    const int bn = blockIdx.x % nstrips;
    const int mbase = bm * 128 + wv * 32;
    const int nbase = bn * 64;

    f32x8 acc[2][4] = {};
    gemm_mainloop(A, W, mbase, nbase, lbuf, acc);

#pragma unroll
    for (int nt = 0; nt < 4; ++nt) {
        const int n = nbase + nt * 16 + ln;
        const float bv = bias[n];
#pragma unroll
        for (int mt = 0; mt < 2; ++mt) {
#pragma unroll
            for (int r = 0; r < 8; ++r) {
                const int m = mbase + mt * 16 + r + 8 * hi;
                C[(size_t)m * N + n] = acc[mt][nt][r] + bv;
            }
        }
    }
}

// -------------------------------------------------------------------- launch
extern "C" void kernel_launch(void* const* d_in, const int* in_sizes, int n_in,
                              void* d_out, int out_size, void* d_ws, size_t ws_size,
                              hipStream_t stream) {
    const int B = 2, S = 4096, D = 1024;
    const int M = B * S;

    size_t off = 0;
    auto alloc = [&](size_t bytes) -> void* {
        void* p = (char*)d_ws + off;
        off = (off + bytes + 255) & ~(size_t)255;
        return p;
    };
    __bf16* xbf  = (__bf16*)alloc((size_t)M * D * 2);
    __bf16* wibf = (__bf16*)alloc((size_t)3 * D * D * 2);
    __bf16* wobf = (__bf16*)alloc((size_t)D * D * 2);
    __bf16* qb   = (__bf16*)alloc((size_t)M * D * 2);
    __bf16* kb   = (__bf16*)alloc((size_t)M * D * 2);
    __bf16* vtb  = (__bf16*)alloc((size_t)M * D * 2);
    __bf16* ob   = (__bf16*)alloc((size_t)M * D * 2);

    for (int path = 0; path < 2; ++path) {
        const float* x  = (const float*)d_in[path == 0 ? 0 : 1];
        const float* wi = (const float*)d_in[path == 0 ? 2 : 6];
        const float* bi = (const float*)d_in[path == 0 ? 3 : 7];
        const float* wo = (const float*)d_in[path == 0 ? 4 : 8];
        const float* bo = (const float*)d_in[path == 0 ? 5 : 9];
        const int wshift = (path == 0) ? 8 : 10;   // window 256 / 1024
        const int window = 1 << wshift;
        float* out = (float*)d_out + (size_t)path * M * D;

        int nx = M * D, nwi = 3 * D * D, nwo = D * D;
        cast_f32_to_bf16<<<(nx / 4 + 255) / 256, 256, 0, stream>>>(x, xbf, nx);
        cast_f32_to_bf16<<<(nwi / 4 + 255) / 256, 256, 0, stream>>>(wi, wibf, nwi);
        cast_f32_to_bf16<<<(nwo / 4 + 255) / 256, 256, 0, stream>>>(wo, wobf, nwo);

        gemm_qkv<<<(M / 128) * (3 * D / 64), 128, 0, stream>>>(
            xbf, wibf, bi, qb, kb, vtb, wshift, 0.125f);

        const int nwin = M / window;
        const int waves = nwin * 16 * (window / 16);
        attn_kernel<<<waves / 4, 128, 0, stream>>>(qb, kb, vtb, ob, wshift);

        gemm_out<<<(M / 128) * (D / 64), 128, 0, stream>>>(ob, wobf, bo, out);
    }
}
